// OutputBlock_72679436583220
// MI455X (gfx1250) — matrix operand order repaired and checked by reading, not compiled, not run on hardware
//
#include <hip/hip_runtime.h>
#include <hip/hip_bf16.h>

// ---------------------------------------------------------------------------
// GemNet-style output block for MI455X (gfx1250).
// bf16 WMMA, LDS-fused residual stacks, register-cached A fragments,
// double-buffered async weight panels (ASYNCcnt) when the toolchain has them.
// ---------------------------------------------------------------------------

typedef __attribute__((ext_vector_type(8)))  __bf16 v8bf;
typedef __attribute__((ext_vector_type(16))) __bf16 v16bf;
typedef __attribute__((ext_vector_type(8)))  float  v8f;
typedef __attribute__((ext_vector_type(4)))  int    v4i;

#define DEVINL __device__ __forceinline__
#define AS1 __attribute__((address_space(1)))
#define AS3 __attribute__((address_space(3)))

#define INV_SQRT2  0.70710678118654752f
#define SILU_SCALE (1.0f / 0.6f)

#if __has_builtin(__builtin_amdgcn_global_load_async_to_lds_b128)
#define HAVE_ASYNC_LDS 1
#else
#define HAVE_ASYNC_LDS 0
#endif

DEVINL float scaled_silu(float v) {
    return v * SILU_SCALE / (1.0f + __expf(-v));
}

DEVINL void wait_async0() {
#if __has_builtin(__builtin_amdgcn_s_wait_asynccnt)
    __builtin_amdgcn_s_wait_asynccnt(0);
#else
    asm volatile("s_wait_asynccnt 0x0" ::: "memory");
#endif
}

// Stage one 16xK bf16 weight panel (contiguous in Wt) into LDS.
template <int K, int BLK>
DEVINL void panel_load(__bf16* __restrict__ dst, const __bf16* __restrict__ src,
                       int tid)
{
    constexpr int CH = (16 * K) / 8;     // 16-byte chunks
#if HAVE_ASYNC_LDS
#pragma unroll
    for (int i = 0; i < CH / BLK; ++i) {
        const int c = tid + i * BLK;
        __builtin_amdgcn_global_load_async_to_lds_b128(
            (AS1 v4i*)(src + (size_t)c * 8),
            (AS3 v4i*)(dst + (size_t)c * 8), 0, 0);
    }
#else
    const uint4* s = (const uint4*)src;
    uint4*       d = (uint4*)dst;
#pragma unroll
    for (int i = 0; i < CH / BLK; ++i)
        d[tid + i * BLK] = s[tid + i * BLK];
    __builtin_prefetch(src + (size_t)16 * K + (size_t)tid * (16 * K / BLK), 0, 0);
#endif
}

// ---------------------------------------------------------------------------
// One dense layer on an LDS-resident tile:
//   sOut[MB x N] = act( sIn[MB x K] @ W[K x N] ),  Wt = W^T [N x K] bf16 global
// MODE 0: out = scaled_silu(acc)
// MODE 1: out = (sComb + scaled_silu(acc)) * INV_SQRT2   (residual combine)
// A fragments for the wave's 16 private rows are cached in VGPRs for the whole
// layer (K/32 x v16bf). Weight panels are double-buffered in sB (2 x 16 x K)
// and fetched with async-to-LDS; one barrier per n-tile.
// ---------------------------------------------------------------------------
template <int K, int N, int MODE, int MB, int BLK>
DEVINL void gemm_layer(const __bf16* __restrict__ sIn,
                       __bf16* __restrict__ sOut,
                       const __bf16* __restrict__ sComb,
                       const __bf16* __restrict__ Wt,
                       __bf16* __restrict__ sB,      // 2 * 16 * K bf16
                       int tid)
{
    static_assert(MB / 16 == BLK / 32, "one wave per 16-row m-tile");
    constexpr int NT = N / 16;
    constexpr int KS = K / 32;
    const int lane = tid & 31;
    const int wid  = tid >> 5;
    const int r0   = wid * 16;
    const int arow = r0 + (lane & 15);
    const int kA   = (lane >> 4) * 8;   // A frag K: {kA..kA+7, kA+16..kA+23}
    const int bcol = lane & 15;
    const int kB   = (lane >> 4) * 8;   // B frag K: {kB..kB+7, kB+16..kB+23}

    // Cache this wave's A fragments (rows are wave-private; same-wave LDS
    // ordering makes the previous layer's epilogue writes visible).
    v16bf aF[KS];
#pragma unroll
    for (int ks = 0; ks < KS; ++ks) {
        const int k0 = ks * 32;
        v8bf alo = *(const v8bf*)(sIn + (size_t)arow * K + k0 + kA);
        v8bf ahi = *(const v8bf*)(sIn + (size_t)arow * K + k0 + kA + 16);
        aF[ks] = __builtin_shufflevector(alo, ahi,
                   0, 1, 2, 3, 4, 5, 6, 7, 8, 9, 10, 11, 12, 13, 14, 15);
    }

    panel_load<K, BLK>(sB, Wt, tid);                 // prologue: panel 0

    for (int nt = 0; nt < NT; ++nt) {
        __bf16* sBcur = sB + (size_t)(nt & 1) * 16 * K;
#if HAVE_ASYNC_LDS
        wait_async0();                               // own async portion done
#endif
        __syncthreads();                             // panel nt visible to all

        if (nt + 1 < NT)                             // overlap next panel fetch
            panel_load<K, BLK>(sB + (size_t)((nt + 1) & 1) * 16 * K,
                               Wt + (size_t)(nt + 1) * 16 * K, tid);

        v8f acc = {};
#pragma unroll
        for (int ks = 0; ks < KS; ++ks) {
            v8bf blo = *(const v8bf*)(sBcur + (size_t)bcol * K + ks * 32 + kB);
            v8bf bhi = *(const v8bf*)(sBcur + (size_t)bcol * K + ks * 32 + kB + 16);
            v16bf b = __builtin_shufflevector(blo, bhi,
                   0, 1, 2, 3, 4, 5, 6, 7, 8, 9, 10, 11, 12, 13, 14, 15);
            acc = __builtin_amdgcn_wmma_f32_16x16x32_bf16(
                      false, aF[ks], false, b, (short)0, acc, false, false);
        }

        // Epilogue (wave-private rows -> no barrier needed before next tile).
        const int crow = r0 + (lane >> 4) * 8;
        const int ccol = nt * 16 + (lane & 15);
#pragma unroll
        for (int r = 0; r < 8; ++r) {
            float y = scaled_silu(acc[r]);
            if (MODE == 1) {
                float xv = (float)sComb[(size_t)(crow + r) * N + ccol];
                y = (xv + y) * INV_SQRT2;
            }
            sOut[(size_t)(crow + r) * N + ccol] = (__bf16)y;
        }
    }
    __syncthreads();
}

// ---------------------------------------------------------------------------
// Transpose + f32->bf16 convert: src = nmat stacked [K x N] row-major,
// dst = nmat stacked [N x K] row-major bf16.
// ---------------------------------------------------------------------------
__global__ __launch_bounds__(256) void convtrans_kernel(
    const float* __restrict__ src, __bf16* __restrict__ dst,
    int K, int N, int total)
{
    int i = blockIdx.x * 256 + threadIdx.x;
    if (i >= total) return;
    int kn  = K * N;
    int mat = i / kn;
    int rem = i - mat * kn;
    int k   = rem / N;
    int n   = rem - k * N;
    dst[(size_t)mat * kn + (size_t)n * K + k] = (__bf16)src[i];
}

// ---------------------------------------------------------------------------
// Edge messages + scatter: xacc[idx[e]] += m[e] * (basis_rad[e] @ W_rbf)
// W_rbf (16x512 f32, 32KB) cached in LDS; f32 atomics land in L2
// (61MB accumulator << 192MB L2).
// ---------------------------------------------------------------------------
__global__ __launch_bounds__(256) void edge_scatter_kernel(
    const float* __restrict__ m, const float* __restrict__ basis_rad,
    const int* __restrict__ idx_atom, const float* __restrict__ W_rbf,
    float* __restrict__ xacc, int nEdges)
{
    constexpr int EB = 32, DE = 512, DR = 16;
    __shared__ __align__(16) float sW[DR * DE];
    __shared__ __align__(16) float sbr[EB * DR];
    __shared__ int sIdx[EB];
    const int tid = threadIdx.x;
    const int e0  = blockIdx.x * EB;

    for (int i = tid; i < DR * DE / 4; i += 256)
        ((float4*)sW)[i] = ((const float4*)W_rbf)[i];
    for (int i = tid; i < EB * DR; i += 256) {
        int e = i / DR;
        sbr[i] = (e0 + e < nEdges) ? basis_rad[(size_t)(e0 + e) * DR + (i % DR)]
                                   : 0.0f;
    }
    if (tid < EB) sIdx[tid] = (e0 + tid < nEdges) ? idx_atom[e0 + tid] : 0;
    __syncthreads();

    for (int j = tid; j < EB * DE; j += 256) {
        int el = j >> 9;            // DE == 512
        int c  = j & (DE - 1);
        if (e0 + el >= nEdges) continue;
        float bemb = 0.0f;
#pragma unroll
        for (int k = 0; k < DR; ++k) bemb += sbr[el * DR + k] * sW[k * DE + c];
        float val = m[(size_t)(e0 + el) * DE + c] * bemb;
        unsafeAtomicAdd(&xacc[(size_t)sIdx[el] * DE + c], val);
    }
}

// ---------------------------------------------------------------------------
// Energy path: xacc*scale -> dense1(512->256)+silu -> 3 residual(256) ->
// (+h)*c -> 2 residual(256) -> out f32. Fused per 64-row tile.
// LDS: 64KB (big/Y) + 32KB (X) + 32KB (double panel) = 128KB.
// ---------------------------------------------------------------------------
__global__ __launch_bounds__(128) void energy_kernel(
    const float* __restrict__ xacc, const float* __restrict__ h,
    const __bf16* __restrict__ WtPreIn, const __bf16* __restrict__ WtPreRes,
    const __bf16* __restrict__ WtE2, const float* __restrict__ scale_sum_p,
    float* __restrict__ outE, int nAtoms)
{
    constexpr int MB = 64, BLK = 128, DE = 512, DA = 256;
    __shared__ __align__(32) __bf16 sBig[MB * DE];   // dense1 input, reused as Y
    __shared__ __align__(32) __bf16 sX[MB * DA];
    __shared__ __align__(32) __bf16 sB[2 * 16 * DE];
    __bf16* sY = sBig;

    const int tid  = threadIdx.x;
    const int row0 = blockIdx.x * MB;
    const float ssum = *scale_sum_p;

    for (int i = tid; i < MB * DE / 4; i += BLK) {
        int r = (i * 4) / DE, c = (i * 4) % DE;
        float4 v = make_float4(0.f, 0.f, 0.f, 0.f);
        if (row0 + r < nAtoms)
            v = *(const float4*)(xacc + (size_t)(row0 + r) * DE + c);
        __bf16* d = sBig + (size_t)r * DE + c;
        d[0] = (__bf16)(v.x * ssum); d[1] = (__bf16)(v.y * ssum);
        d[2] = (__bf16)(v.z * ssum); d[3] = (__bf16)(v.w * ssum);
    }
    __syncthreads();

    gemm_layer<DE, DA, 0, MB, BLK>(sBig, sX, nullptr, WtPreIn, sB, tid);

    for (int l = 0; l < 3; ++l) {
        gemm_layer<DA, DA, 0, MB, BLK>(sX, sY, nullptr,
            WtPreRes + (size_t)(2 * l) * DA * DA, sB, tid);
        gemm_layer<DA, DA, 1, MB, BLK>(sY, sX, sX,
            WtPreRes + (size_t)(2 * l + 1) * DA * DA, sB, tid);
    }

    for (int i = tid; i < MB * DA; i += BLK) {   // x = (x + h) * inv_sqrt2
        int r = i / DA, c = i % DA;
        float xv = (float)sX[i];
        if (row0 + r < nAtoms)
            xv = (xv + h[(size_t)(row0 + r) * DA + c]) * INV_SQRT2;
        sX[i] = (__bf16)xv;
    }
    __syncthreads();

    for (int l = 0; l < 2; ++l) {
        gemm_layer<DA, DA, 0, MB, BLK>(sX, sY, nullptr,
            WtE2 + (size_t)(2 * l) * DA * DA, sB, tid);
        gemm_layer<DA, DA, 1, MB, BLK>(sY, sX, sX,
            WtE2 + (size_t)(2 * l + 1) * DA * DA, sB, tid);
    }

    for (int i = tid; i < MB * DA; i += BLK) {
        int r = i / DA, c = i % DA;
        if (row0 + r < nAtoms)
            outE[(size_t)(row0 + r) * DA + c] = (float)sX[i];
    }
}

// ---------------------------------------------------------------------------
// Force path: m -> 3 residual(512) -> * (basis_rad @ W_rbf_F) * scale -> out.
// LDS: 64KB + 64KB + 32KB = 160KB.
// ---------------------------------------------------------------------------
__global__ __launch_bounds__(128) void force_kernel(
    const float* __restrict__ m, const float* __restrict__ basis_rad,
    const float* __restrict__ W_rbf_F, const __bf16* __restrict__ WtF,
    const float* __restrict__ scale_p, float* __restrict__ outF, int nEdges)
{
    constexpr int MB = 64, BLK = 128, DE = 512, DR = 16;
    __shared__ __align__(32) __bf16 sX[MB * DE];
    __shared__ __align__(32) __bf16 sY[MB * DE];
    __shared__ __align__(32) __bf16 sB[2 * 16 * DE];

    const int tid  = threadIdx.x;
    const int row0 = blockIdx.x * MB;

    for (int i = tid; i < MB * DE / 4; i += BLK) {
        int r = (i * 4) / DE, c = (i * 4) % DE;
        float4 v = make_float4(0.f, 0.f, 0.f, 0.f);
        if (row0 + r < nEdges)
            v = *(const float4*)(m + (size_t)(row0 + r) * DE + c);
        __bf16* d = sX + (size_t)r * DE + c;
        d[0] = (__bf16)v.x; d[1] = (__bf16)v.y;
        d[2] = (__bf16)v.z; d[3] = (__bf16)v.w;
    }
    __syncthreads();

    for (int l = 0; l < 3; ++l) {
        gemm_layer<DE, DE, 0, MB, BLK>(sX, sY, nullptr,
            WtF + (size_t)(2 * l) * DE * DE, sB, tid);
        gemm_layer<DE, DE, 1, MB, BLK>(sY, sX, sX,
            WtF + (size_t)(2 * l + 1) * DE * DE, sB, tid);
    }

    // Epilogue: recompute basis embedding (16 MACs/elem) instead of storing a
    // second 614MB tensor. Reuse sY (32KB of 64KB) for W_rbf_F, sB for rad.
    float* sW  = (float*)sY;   // 16*512 f32 = 32KB
    float* sBR = (float*)sB;   // 64*16  f32 =  4KB
    for (int i = tid; i < DR * DE / 4; i += BLK)
        ((float4*)sW)[i] = ((const float4*)W_rbf_F)[i];
    for (int i = tid; i < MB * DR; i += BLK) {
        int r = i / DR;
        sBR[i] = (row0 + r < nEdges)
                     ? basis_rad[(size_t)(row0 + r) * DR + (i % DR)] : 0.0f;
    }
    __syncthreads();

    const float sc = *scale_p;
    for (int j = tid; j < MB * DE; j += BLK) {
        int r = j >> 9;            // DE == 512
        int c = j & (DE - 1);
        if (row0 + r >= nEdges) continue;
        float bemb = 0.0f;
#pragma unroll
        for (int k = 0; k < DR; ++k) bemb += sBR[r * DR + k] * sW[k * DE + c];
        outF[(size_t)(row0 + r) * DE + c] =
            (float)sX[(size_t)r * DE + c] * bemb * sc;
    }
}

// ---------------------------------------------------------------------------
extern "C" void kernel_launch(void* const* d_in, const int* in_sizes, int n_in,
                              void* d_out, int out_size, void* d_ws,
                              size_t ws_size, hipStream_t stream)
{
    (void)n_in; (void)out_size; (void)ws_size;
    const int DA = 256, DE = 512;

    const float* h          = (const float*)d_in[0];
    const float* m          = (const float*)d_in[1];
    const float* br         = (const float*)d_in[2];
    const int*   idx        = (const int*)d_in[3];
    const float* W_rbf      = (const float*)d_in[4];
    const float* W_rbf_F    = (const float*)d_in[5];
    const float* W_pre_in   = (const float*)d_in[6];
    const float* W_pre_res  = (const float*)d_in[7];
    const float* W_e2_res   = (const float*)d_in[8];
    const float* W_F_res    = (const float*)d_in[9];
    const float* scale_sum  = (const float*)d_in[10];
    const float* scale_rbfF = (const float*)d_in[11];

    const int nAtoms = in_sizes[0] / DA;
    const int nEdges = in_sizes[3];

    char*  ws  = (char*)d_ws;
    size_t off = 0;
    float*  xacc     = (float*)(ws + off);  off += (size_t)nAtoms * DE * 4;
    __bf16* WtPreIn  = (__bf16*)(ws + off); off += (size_t)DE * DA * 2;
    __bf16* WtPreRes = (__bf16*)(ws + off); off += (size_t)6 * DA * DA * 2;
    __bf16* WtE2     = (__bf16*)(ws + off); off += (size_t)4 * DA * DA * 2;
    __bf16* WtF      = (__bf16*)(ws + off); off += (size_t)6 * DE * DE * 2;

    float* outE = (float*)d_out;
    float* outF = outE + (size_t)nAtoms * DA;

    (void)hipMemsetAsync(xacc, 0, (size_t)nAtoms * DE * 4, stream);

    { int t = DE * DA;
      convtrans_kernel<<<(t + 255) / 256, 256, 0, stream>>>(W_pre_in, WtPreIn, DE, DA, t); }
    { int t = 6 * DA * DA;
      convtrans_kernel<<<(t + 255) / 256, 256, 0, stream>>>(W_pre_res, WtPreRes, DA, DA, t); }
    { int t = 4 * DA * DA;
      convtrans_kernel<<<(t + 255) / 256, 256, 0, stream>>>(W_e2_res, WtE2, DA, DA, t); }
    { int t = 6 * DE * DE;
      convtrans_kernel<<<(t + 255) / 256, 256, 0, stream>>>(W_F_res, WtF, DE, DE, t); }

    edge_scatter_kernel<<<(nEdges + 31) / 32, 256, 0, stream>>>(
        m, br, idx, W_rbf, xacc, nEdges);

    energy_kernel<<<(nAtoms + 63) / 64, 128, 0, stream>>>(
        xacc, h, WtPreIn, WtPreRes, WtE2, scale_sum, outE, nAtoms);

    force_kernel<<<(nEdges + 63) / 64, 128, 0, stream>>>(
        m, br, W_rbf_F, WtF, scale_rbfF, outF, nEdges);
}
